// Decoder_35983236005935
// MI455X (gfx1250) — matrix operand order repaired and checked by reading, not compiled, not run on hardware
//
#include <hip/hip_runtime.h>

#define B_ 32
#define T_ 150
#define L_ 400
#define FEAT_ 512
#define RNN_ 512
#define EMB_ 256
#define ATT_ 512
#define VOCAB_ 5000

typedef __attribute__((ext_vector_type(16))) __bf16 v16bf;
typedef __attribute__((ext_vector_type(8)))  __bf16 v8bf;
typedef __attribute__((ext_vector_type(8)))  float  v8f;

// ---------- helpers ----------
__device__ __forceinline__ __bf16 f2b(float f) {
  unsigned u; __builtin_memcpy(&u, &f, 4);
  unsigned r = u + 0x7FFFu + ((u >> 16) & 1u);   // round-to-nearest-even
  unsigned short h = (unsigned short)(r >> 16);
  __bf16 b; __builtin_memcpy(&b, &h, 2);
  return b;
}

// ---------- fp32 -> bf16 elementwise ----------
__global__ void cvt_bf16_kernel(const float* __restrict__ src,
                                __bf16* __restrict__ dst, int n) {
  int i = blockIdx.x * blockDim.x + threadIdx.x;
  if (i < n) dst[i] = f2b(src[i]);
}

// ---------- pack fp32 B[K,N] into WMMA bf16 B-fragment layout ----------
// Bp[((kt*tilesN + nt)*32 + lane)*16 + j]  holds  B[kt*32 + 16*(lane>>4) + j][nt*16 + (lane&15)]
// N-edge columns are zero-padded so the GEMM loop needs no predication.
__global__ void pack_b_kernel(const float* __restrict__ B, int ldb,
                              int N, int K, int tilesN,
                              __bf16* __restrict__ Bp) {
  int idx = blockIdx.x * blockDim.x + threadIdx.x;
  int total = (K >> 5) * tilesN * 512;
  if (idx >= total) return;
  int j    = idx & 15;
  int lane = (idx >> 4) & 31;
  int t    = idx >> 9;
  int nt   = t % tilesN;
  int kt   = t / tilesN;
  int k = kt * 32 + 16 * (lane >> 4) + j;
  int n = nt * 16 + (lane & 15);
  float v = (n < N) ? B[(size_t)k * ldb + n] : 0.0f;
  Bp[idx] = f2b(v);
}

// ---------- bf16 WMMA GEMM with pre-packed B fragments ----------
// C[M,N] = A[M,K] x B[K,N]; A row-major bf16 (lda mult of 16),
// Bp packed fragments; C fp32 (nullable), Cb bf16 (nullable); act 1 => tanh.
// one wave -> one 16x16 tile; K multiple of 64; M multiple of 16.
__global__ void gemm_bf16_wmma(const __bf16* __restrict__ A, int lda,
                               const __bf16* __restrict__ Bp,
                               float* __restrict__ C, int ldc,
                               __bf16* __restrict__ Cb, int ldcb,
                               int M, int N, int K, int tilesN, int act) {
  int wv   = threadIdx.x >> 5;
  int lane = threadIdx.x & 31;
  int tile = blockIdx.x * (blockDim.x >> 5) + wv;
  if (tile >= (M >> 4) * tilesN) return;
  int m0   = (tile / tilesN) << 4;
  int tn   = tile % tilesN;
  int n0   = tn << 4;
  int half = lane >> 4;
  int l16  = lane & 15;

  const __bf16* Arow  = A + (size_t)(m0 + l16) * lda + 16 * half;
  const __bf16* Bfrag = Bp + ((size_t)tn * 32 + lane) * 16;
  const size_t  bstep = (size_t)tilesN * 512;   // packed elems per K-tile

  v8f acc0 = {}, acc1 = {};
  int ktiles = K >> 5;                          // even (K % 64 == 0)
  for (int kt = 0; kt < ktiles; kt += 2) {
    // A fragments: two aligned 16B loads each
    v8bf a0lo = *(const v8bf*)(Arow + kt * 32);
    v8bf a0hi = *(const v8bf*)(Arow + kt * 32 + 8);
    v8bf a1lo = *(const v8bf*)(Arow + kt * 32 + 32);
    v8bf a1hi = *(const v8bf*)(Arow + kt * 32 + 40);
    v16bf a0 = __builtin_shufflevector(a0lo, a0hi, 0,1,2,3,4,5,6,7,8,9,10,11,12,13,14,15);
    v16bf a1 = __builtin_shufflevector(a1lo, a1hi, 0,1,2,3,4,5,6,7,8,9,10,11,12,13,14,15);
    // B fragments: contiguous aligned 32B per lane
    v16bf b0 = *(const v16bf*)(Bfrag + (size_t)kt * bstep);
    v16bf b1 = *(const v16bf*)(Bfrag + (size_t)(kt + 1) * bstep);
    acc0 = __builtin_amdgcn_wmma_f32_16x16x32_bf16(
        false, a0, false, b0, (short)0, acc0, false, false);
    acc1 = __builtin_amdgcn_wmma_f32_16x16x32_bf16(
        false, a1, false, b1, (short)0, acc1, false, false);
  }

  int n = n0 + l16;
  if (n < N) {
    #pragma unroll
    for (int v = 0; v < 8; ++v) {
      int   row = m0 + v + 8 * half;
      float val = acc0[v] + acc1[v];
      if (act == 1) val = tanhf(val);
      if (C)  C[(size_t)row * ldc + n]   = val;
      if (Cb) Cb[(size_t)row * ldcb + n] = f2b(val);
    }
  }
}

// ---------- init carry ----------
__global__ void init_kernel(const float* __restrict__ init_state,
                            float* __restrict__ state,
                            __bf16* __restrict__ state_bf,
                            __bf16* __restrict__ prev_out) {
  int idx = blockIdx.x * blockDim.x + threadIdx.x;
  if (idx >= B_ * RNN_) return;
  float v = init_state[idx];
  state[idx]    = v;
  state_bf[idx] = f2b(v);
  prev_out[idx] = f2b(0.0f);
}

// ---------- inp = concat(emb[tok], last_out) (bf16) ----------
__global__ void embed_concat_kernel(const __bf16* __restrict__ emb,
                                    const int* __restrict__ formula, int t,
                                    const __bf16* __restrict__ prev_out,
                                    __bf16* __restrict__ inp) {
  int idx = blockIdx.x * blockDim.x + threadIdx.x;
  if (idx >= B_ * (EMB_ + RNN_)) return;
  int b = idx / (EMB_ + RNN_);
  int j = idx % (EMB_ + RNN_);
  __bf16 v;
  if (j < EMB_) {
    int tok = formula[b * T_ + t];
    v = emb[(size_t)tok * EMB_ + j];
  } else {
    v = prev_out[b * RNN_ + (j - EMB_)];
  }
  inp[idx] = v;
}

// ---------- GRU gates (reset_after=true, order z,r,h) ----------
__global__ void gru_gate_kernel(const float* __restrict__ mx,
                                const float* __restrict__ mh,
                                const float* __restrict__ gbias,
                                float* __restrict__ state,
                                __bf16* __restrict__ state_bf,
                                __bf16* __restrict__ hc) {
  int idx = blockIdx.x * blockDim.x + threadIdx.x;
  if (idx >= B_ * RNN_) return;
  int b = idx / RNN_;
  int i = idx % RNN_;
  const float* mxb = mx + (size_t)b * 3 * RNN_;
  const float* mhb = mh + (size_t)b * 3 * RNN_;
  const float* b0  = gbias;
  const float* b1  = gbias + 3 * RNN_;
  float xz = mxb[i]            + b0[i];
  float xr = mxb[RNN_ + i]     + b0[RNN_ + i];
  float xh = mxb[2 * RNN_ + i] + b0[2 * RNN_ + i];
  float rz = mhb[i]            + b1[i];
  float rr = mhb[RNN_ + i]     + b1[RNN_ + i];
  float rh = mhb[2 * RNN_ + i] + b1[2 * RNN_ + i];
  float z  = 1.0f / (1.0f + __expf(-(xz + rz)));
  float r  = 1.0f / (1.0f + __expf(-(xr + rr)));
  float hh = tanhf(xh + r * rh);
  float h  = z * state[idx] + (1.0f - z) * hh;
  state[idx]    = h;
  state_bf[idx] = f2b(h);
  hc[(size_t)b * (RNN_ + FEAT_) + i] = f2b(h);
}

// ---------- score[b,l] = tanh(feat_proj[b,l,:] + q[b,:]) . V ----------
__global__ void attn_score_kernel(const float* __restrict__ fp,
                                  const float* __restrict__ q,
                                  const float* __restrict__ V,
                                  float* __restrict__ score) {
  int wv   = threadIdx.x >> 5;
  int lane = threadIdx.x & 31;
  int bl   = blockIdx.x * (blockDim.x >> 5) + wv;
  if (bl >= B_ * L_) return;
  int b = bl / L_;
  const float* fr = fp + (size_t)bl * ATT_;
  const float* qr = q + (size_t)b * ATT_;
  float p = 0.0f;
  for (int c = lane; c < ATT_; c += 32)
    p += tanhf(fr[c] + qr[c]) * V[c];
  #pragma unroll
  for (int off = 16; off > 0; off >>= 1)
    p += __shfl_xor(p, off, 32);
  if (lane == 0) score[bl] = p;
}

// ---------- softmax over L per batch row (in-place) ----------
__global__ void softmax_kernel(float* __restrict__ score) {
  __shared__ float red[256];
  int b = blockIdx.x, tid = threadIdx.x;
  float* s = score + (size_t)b * L_;
  float m = -1e30f;
  for (int l = tid; l < L_; l += 256) m = fmaxf(m, s[l]);
  red[tid] = m; __syncthreads();
  for (int o = 128; o > 0; o >>= 1) {
    if (tid < o) red[tid] = fmaxf(red[tid], red[tid + o]);
    __syncthreads();
  }
  float mx = red[0]; __syncthreads();
  float sum = 0.0f;
  for (int l = tid; l < L_; l += 256) { float e = __expf(s[l] - mx); s[l] = e; sum += e; }
  red[tid] = sum; __syncthreads();
  for (int o = 128; o > 0; o >>= 1) {
    if (tid < o) red[tid] += red[tid + o];
    __syncthreads();
  }
  float inv = 1.0f / red[0];
  for (int l = tid; l < L_; l += 256) s[l] *= inv;
}

// ---------- ctx[b,f] = sum_l w[b,l]*features[b,l,f] -> hc[:,512:1024] bf16 ----------
__global__ void ctx_kernel(const float* __restrict__ w,
                           const float* __restrict__ feats,
                           __bf16* __restrict__ hc) {
  int idx = blockIdx.x * blockDim.x + threadIdx.x;
  if (idx >= B_ * FEAT_) return;
  int b = idx / FEAT_;
  int f = idx % FEAT_;
  const float* wb = w + (size_t)b * L_;
  const float* fb = feats + (size_t)b * L_ * FEAT_ + f;
  float acc = 0.0f;
  for (int l = 0; l < L_; ++l) acc += wb[l] * fb[(size_t)l * FEAT_];
  hc[(size_t)b * (RNN_ + FEAT_) + RNN_ + f] = f2b(acc);
}

// ---------- host ----------
extern "C" void kernel_launch(void* const* d_in, const int* in_sizes, int n_in,
                              void* d_out, int out_size, void* d_ws, size_t ws_size,
                              hipStream_t stream) {
  const float* features   = (const float*)d_in[0];
  const float* init_state = (const float*)d_in[1];
  const float* emb        = (const float*)d_in[2];
  const float* gruK       = (const float*)d_in[3];
  const float* gruR       = (const float*)d_in[4];
  const float* gbias      = (const float*)d_in[5];
  const float* W1         = (const float*)d_in[6];
  const float* W2         = (const float*)d_in[7];
  const float* Vv         = (const float*)d_in[8];
  const float* outW       = (const float*)d_in[9];
  const float* projW      = (const float*)d_in[10];
  const int*   formula    = (const int*)d_in[11];
  float*       out        = (float*)d_out;

  auto cdiv = [](int a, int b) { return (a + b - 1) / b; };

  char* p = (char*)(((uintptr_t)d_ws + 255) & ~(uintptr_t)255);
  auto alloc = [&](size_t bytes) {
    void* r = (void*)p;
    p += (bytes + 255) & ~(size_t)255;
    return r;
  };
  auto packed_elems = [&](int N, int K) {
    return (size_t)(K >> 5) * cdiv(N, 16) * 512;
  };

  // bf16 A-side staging
  __bf16* feats_bf = (__bf16*)alloc(2ull * B_ * L_ * FEAT_);
  __bf16* emb_bf   = (__bf16*)alloc(2ull * VOCAB_ * EMB_);
  __bf16* inp_bf   = (__bf16*)alloc(2ull * B_ * (EMB_ + RNN_));
  __bf16* state_bf = (__bf16*)alloc(2ull * B_ * RNN_);
  __bf16* hc_bf    = (__bf16*)alloc(2ull * B_ * (RNN_ + FEAT_));
  __bf16* prev_bf  = (__bf16*)alloc(2ull * B_ * RNN_);
  // packed bf16 B fragments
  __bf16* gruK_pk  = (__bf16*)alloc(2ull * packed_elems(3 * RNN_, EMB_ + RNN_));
  __bf16* gruR_pk  = (__bf16*)alloc(2ull * packed_elems(3 * RNN_, RNN_));
  __bf16* W1_pk    = (__bf16*)alloc(2ull * packed_elems(ATT_, FEAT_));
  __bf16* W2_pk    = (__bf16*)alloc(2ull * packed_elems(ATT_, RNN_));
  __bf16* outW_pk  = (__bf16*)alloc(2ull * packed_elems(RNN_, RNN_ + FEAT_));
  __bf16* projW_pk = (__bf16*)alloc(2ull * packed_elems(VOCAB_, RNN_));
  // fp32 scratch
  float* feat_proj = (float*)alloc(4ull * B_ * L_ * ATT_);
  float* mx        = (float*)alloc(4ull * B_ * 3 * RNN_);
  float* mh        = (float*)alloc(4ull * B_ * 3 * RNN_);
  float* state_f   = (float*)alloc(4ull * B_ * RNN_);
  float* q         = (float*)alloc(4ull * B_ * ATT_);
  float* score     = (float*)alloc(4ull * B_ * L_);

  auto cvt = [&](const float* s, __bf16* d, int n) {
    cvt_bf16_kernel<<<cdiv(n, 256), 256, 0, stream>>>(s, d, n);
  };
  auto pack = [&](const float* Bm, int N, int K, __bf16* Bp) {
    int tilesN = cdiv(N, 16);
    int total  = (K >> 5) * tilesN * 512;
    pack_b_kernel<<<cdiv(total, 256), 256, 0, stream>>>(Bm, N, N, K, tilesN, Bp);
  };
  auto gemm = [&](const __bf16* A, int lda, const __bf16* Bp,
                  float* C, int ldc, __bf16* Cb, int ldcb,
                  int M, int N, int K, int act) {
    int tilesN = cdiv(N, 16);
    int tiles  = (M >> 4) * tilesN;
    gemm_bf16_wmma<<<cdiv(tiles, 4), 128, 0, stream>>>(
        A, lda, Bp, C, ldc, Cb, ldcb, M, N, K, tilesN, act);
  };

  // ---- prologue: convert A-side data, pack all weight matrices ----
  cvt(features, feats_bf, B_ * L_ * FEAT_);
  cvt(emb,      emb_bf,   VOCAB_ * EMB_);
  pack(gruK,  3 * RNN_, EMB_ + RNN_, gruK_pk);
  pack(gruR,  3 * RNN_, RNN_,        gruR_pk);
  pack(W1,    ATT_,     FEAT_,       W1_pk);
  pack(W2,    ATT_,     RNN_,        W2_pk);
  pack(outW,  RNN_,     RNN_ + FEAT_, outW_pk);
  pack(projW, VOCAB_,   RNN_,        projW_pk);
  init_kernel<<<cdiv(B_ * RNN_, 256), 256, 0, stream>>>(
      init_state, state_f, state_bf, prev_bf);
  // feat_proj = features @ W1   [12800 x 512]
  gemm(feats_bf, FEAT_, W1_pk, feat_proj, ATT_, nullptr, 0,
       B_ * L_, ATT_, FEAT_, 0);

  // ---- 150 sequential decode steps ----
  for (int t = 0; t < T_; ++t) {
    embed_concat_kernel<<<cdiv(B_ * (EMB_ + RNN_), 256), 256, 0, stream>>>(
        emb_bf, formula, t, prev_bf, inp_bf);
    gemm(inp_bf, EMB_ + RNN_, gruK_pk, mx, 3 * RNN_, nullptr, 0,
         B_, 3 * RNN_, EMB_ + RNN_, 0);
    gemm(state_bf, RNN_, gruR_pk, mh, 3 * RNN_, nullptr, 0,
         B_, 3 * RNN_, RNN_, 0);
    gru_gate_kernel<<<cdiv(B_ * RNN_, 256), 256, 0, stream>>>(
        mx, mh, gbias, state_f, state_bf, hc_bf);
    gemm(state_bf, RNN_, W2_pk, q, ATT_, nullptr, 0,
         B_, ATT_, RNN_, 0);
    attn_score_kernel<<<cdiv(B_ * L_, 4), 128, 0, stream>>>(
        feat_proj, q, Vv, score);
    softmax_kernel<<<B_, 256, 0, stream>>>(score);
    ctx_kernel<<<cdiv(B_ * FEAT_, 256), 256, 0, stream>>>(
        score, features, hc_bf);
    gemm(hc_bf, RNN_ + FEAT_, outW_pk, nullptr, 0, prev_bf, RNN_,
         B_, RNN_, RNN_ + FEAT_, 1);                     // tanh, bf16 out
    gemm(prev_bf, RNN_, projW_pk, out + (size_t)t * VOCAB_, T_ * VOCAB_,
         nullptr, 0, B_, VOCAB_, RNN_, 0);               // logits
  }
}
